// RoutingLayer_4449586119503
// MI455X (gfx1250) — matrix operand order repaired and checked by reading, not compiled, not run on hardware
//
#include <hip/hip_runtime.h>
#include <math.h>

// ---- problem constants (match reference) ----
#define N_PTS  50000
#define M_NB   32
#define D_FEAT 128
#define K_CAP  8
#define DD     16
#define EPS_NRM 1e-12f

#define NW 4                 // waves (points) per block
#define ZS_STRIDE 136        // halves per zs row: 128 + 8 pad (16B-aligned, bank-spread)

typedef _Float16 v16h __attribute__((ext_vector_type(16)));
typedef _Float16 v8h  __attribute__((ext_vector_type(8)));
typedef _Float16 v4h  __attribute__((ext_vector_type(4)));
typedef float    v8f  __attribute__((ext_vector_type(8)));
typedef float    v4f  __attribute__((ext_vector_type(4)));

static __device__ __forceinline__ v8f wmma16x16x32(v16h a, v16h b, v8f c) {
  // D = A(16x32,f16) * B(32x16,f16) + C(16x16,f32)
  return __builtin_amdgcn_wmma_f32_16x16x32_f16(false, a, false, b, (short)0, c,
                                                false, false);
}

__global__ __launch_bounds__(NW * 32) void routing_kernel(
    const float* __restrict__ x,      // [N, 128] f32
    const int*   __restrict__ nbr,    // [N*32] indices in [0, N] (N = pad row)
    const int*   __restrict__ miter,  // [1] max_iter
    float*       __restrict__ out)    // [N, 128] f32
{
  __shared__ _Float16 zs_all[NW][M_NB * ZS_STRIDE];

  const int lane  = threadIdx.x & 31;
  const int wave  = threadIdx.x >> 5;
  const int klane = lane & 15;          // column index in WMMA B/D layouts
  const int hi    = lane >> 4;          // 0: lanes 0-15, 1: lanes 16-31

  int n = blockIdx.x * NW + wave;
  const bool valid = (n < N_PTS);
  if (!valid) n = N_PTS - 1;            // clamp; keep wave alive for barriers

  _Float16* zsw = zs_all[wave];
  const int max_iter = miter[0];

  // ================= gather + per-capsule normalize neighbors -> zs (f16) ===
  // Per m: 32 lanes load one 512B row coalesced (float4/lane), normalize each
  // 16-float capsule (4-lane groups) and store 4 halves (ds_store_b64).
  const int myIdx = nbr[n * M_NB + lane];
  #pragma unroll 4
  for (int m = 0; m < M_NB; ++m) {
    const int row = __shfl(myIdx, m, 32);         // uniform across wave
    v4f v;
    if (row < N_PTS) {
      v = *(const v4f*)(x + (size_t)row * D_FEAT + 4 * lane);
    } else {
      v = (v4f){0.f, 0.f, 0.f, 0.f};              // zero pad row
    }
    float ss = v[0]*v[0] + v[1]*v[1] + v[2]*v[2] + v[3]*v[3];
    ss += __shfl_xor(ss, 1, 32);                  // sum within 4-lane capsule
    ss += __shfl_xor(ss, 2, 32);
    const float sc = 1.0f / fmaxf(sqrtf(ss), EPS_NRM);
    v4h hv;
    hv[0] = (_Float16)(v[0] * sc); hv[1] = (_Float16)(v[1] * sc);
    hv[2] = (_Float16)(v[2] * sc); hv[3] = (_Float16)(v[3] * sc);
    *(v4h*)(zsw + m * ZS_STRIDE + 4 * lane) = hv;
  }

  // ================= x_c: normalized self features, register layout =========
  // lane holds x_c[k = klane&7][d = r + 8*hi], r = 0..7
  const int kc = klane & 7;                       // duplicate for klane>=8 (unused)
  float xc[8];
  {
    const float* xp = x + (size_t)n * D_FEAT + kc * DD + 8 * hi;
    v4f a = *(const v4f*)xp;
    v4f b = *(const v4f*)(xp + 4);
    float ss = a[0]*a[0]+a[1]*a[1]+a[2]*a[2]+a[3]*a[3]
             + b[0]*b[0]+b[1]*b[1]+b[2]*b[2]+b[3]*b[3];
    ss += __shfl_xor(ss, 16, 32);                 // other 8 dims in partner lane
    const float sc = 1.0f / fmaxf(sqrtf(ss), EPS_NRM);
    xc[0]=a[0]*sc; xc[1]=a[1]*sc; xc[2]=a[2]*sc; xc[3]=a[3]*sc;
    xc[4]=b[0]*sc; xc[5]=b[1]*sc; xc[6]=b[2]*sc; xc[7]=b[3]*sc;
  }

  __syncthreads();   // zs stores visible before transposed reads

  // ================= Z^T A-tiles in registers (built once) ==================
  // Au[t2]: A(16x32) with rows = dims 16*t2..16*t2+15 (lane row = klane),
// K-index = m, contiguous: m = h + 16*hi.
  v16h Au[8];
  #pragma unroll
  for (int t2 = 0; t2 < 8; ++t2) {
    #pragma unroll
    for (int h = 0; h < 16; ++h) {
      const int m = h + 16 * hi;
      Au[t2][h] = zsw[m * ZS_STRIDE + 16 * t2 + klane];
    }
  }

  // ================= routing iterations =====================================
  float u[8];
  #pragma unroll
  for (int r = 0; r < 8; ++r) u[r] = 0.0f;

  v16h Bu;                                   // P in B(32x16) layout (f16)
  #pragma unroll
  for (int h = 0; h < 16; ++h) Bu[h] = (_Float16)0.125f;   // softmax(0) = 1/8

  for (int it = 0; it < max_iter; ++it) {
    if (it > 0) {
      // ---- p = einsum('mkd,kd->mk', z, u) via block-diagonal Ubig ----
      float uX[8];
      #pragma unroll
      for (int r = 0; r < 8; ++r) uX[r] = __shfl_xor(u[r], 16, 32);

      v8f pacc[2];
      #pragma unroll
      for (int t = 0; t < 2; ++t)
        #pragma unroll
        for (int r = 0; r < 8; ++r) pacc[t][r] = 0.0f;

      #pragma unroll
      for (int c = 0; c < 4; ++c) {          // K-chunk c: dims 32c..32c+31 (capsules 2c,2c+1)
        // B layout: col = klane, K = h + 16*hi. Nonzero col k' = 2c + hi.
        v16h Bp;
        #pragma unroll
        for (int h = 0; h < 16; ++h) Bp[h] = (_Float16)0.0f;
        const bool sel = (!hi && klane == 2 * c) || (hi && klane == 2 * c + 1);
        if (sel) {
          #pragma unroll
          for (int h = 0; h < 8; ++h) {
            Bp[h]     = (_Float16)(hi ? uX[h] : u[h]);   // d' = h
            Bp[h + 8] = (_Float16)(hi ? u[h] : uX[h]);   // d' = h+8
          }
        }
        #pragma unroll
        for (int t = 0; t < 2; ++t) {
          // A tile: rows m = 16t+klane, dims 32c + 16*hi + 0..15
          const _Float16* ap = zsw + (16 * t + klane) * ZS_STRIDE + 32 * c + 16 * hi;
          const v8h lo = *(const v8h*)ap;          // ds_load_b128
          const v8h up = *(const v8h*)(ap + 8);    // ds_load_b128
          v16h A;
          #pragma unroll
          for (int h = 0; h < 8; ++h) { A[h] = lo[h]; A[h + 8] = up[h]; }
          pacc[t] = wmma16x16x32(A, Bp, pacc[t]);
        }
      }

      // ---- softmax over k (= lane&15, groups of 8 lanes) ----
      #pragma unroll
      for (int t = 0; t < 2; ++t) {
        #pragma unroll
        for (int r = 0; r < 8; ++r) {
          float v = pacc[t][r];
          float mx = v;
          mx = fmaxf(mx, __shfl_xor(mx, 1, 32));
          mx = fmaxf(mx, __shfl_xor(mx, 2, 32));
          mx = fmaxf(mx, __shfl_xor(mx, 4, 32));
          const float e = __expf(v - mx);
          float s = e;
          s += __shfl_xor(s, 1, 32);
          s += __shfl_xor(s, 2, 32);
          s += __shfl_xor(s, 4, 32);
          pacc[t][r] = e / s;
        }
      }

      // ---- repack P (D layout) into B layout with lane^16 exchange ----
      float q0[8], q1[8];
      #pragma unroll
      for (int r = 0; r < 8; ++r) {
        q0[r] = __shfl_xor(pacc[0][r], 16, 32);
        q1[r] = __shfl_xor(pacc[1][r], 16, 32);
      }
      #pragma unroll
      for (int h = 0; h < 8; ++h) {
        Bu[h]     = (_Float16)(hi ? q1[h] : pacc[0][h]);  // m = h + 16*hi
        Bu[h + 8] = (_Float16)(hi ? pacc[1][h] : q0[h]);  // m = h+8 + 16*hi
      }
    }

    // ---- u = einsum('mkd,mk->kd', z, p) + x_c : 8 WMMAs, diagonal extract ----
    #pragma unroll
    for (int t2 = 0; t2 < 8; ++t2) {
      v8f acc;
      #pragma unroll
      for (int r = 0; r < 8; ++r) acc[r] = 0.0f;
      acc = wmma16x16x32(Au[t2], Bu, acc);
      if (klane == t2) {                 // column k == capsule t2 is the real data
        #pragma unroll
        for (int r = 0; r < 8; ++r) u[r] = acc[r] + xc[r];
      }
    }

    // ---- per-capsule normalize (not on last iteration) ----
    if (it + 1 < max_iter) {
      float ss = 0.0f;
      #pragma unroll
      for (int r = 0; r < 8; ++r) ss += u[r] * u[r];
      ss += __shfl_xor(ss, 16, 32);
      const float sc = 1.0f / fmaxf(sqrtf(ss), EPS_NRM);
      #pragma unroll
      for (int r = 0; r < 8; ++r) u[r] *= sc;
    }
  }

  // ================= store u: coalesced, covers all 128 dims ================
  if (valid && klane < 8) {
    float* op = out + (size_t)n * D_FEAT + klane * DD + 8 * hi;
    v4f a, b;
    a[0]=u[0]; a[1]=u[1]; a[2]=u[2]; a[3]=u[3];
    b[0]=u[4]; b[1]=u[5]; b[2]=u[6]; b[3]=u[7];
    *(v4f*)op = a;
    *(v4f*)(op + 4) = b;
  }
}

extern "C" void kernel_launch(void* const* d_in, const int* in_sizes, int n_in,
                              void* d_out, int out_size, void* d_ws, size_t ws_size,
                              hipStream_t stream) {
  (void)in_sizes; (void)n_in; (void)out_size; (void)d_ws; (void)ws_size;
  const float* x     = (const float*)d_in[0];
  const int*   nbr   = (const int*)d_in[1];   // int32 per harness convention
  const int*   miter = (const int*)d_in[2];
  float*       out   = (float*)d_out;

  const int blocks = (N_PTS + NW - 1) / NW;   // 12500 blocks of 128 threads (4 waves)
  routing_kernel<<<blocks, NW * 32, 0, stream>>>(x, nbr, miter, out);
}
